// RotEncoderFusion_78417512891254
// MI455X (gfx1250) — matrix operand order repaired and checked by reading, not compiled, not run on hardware
//
#include <hip/hip_runtime.h>
#include <hip/hip_bf16.h>

// Problem constants (match reference)
#define BB    2
#define NN    2048
#define MMN   64          // SOM nodes
#define KTOP  3
#define KN    (KTOP*NN)   // 6144 stacked points
#define RRC   12          // rotation copies
#define BRC   (BB*RRC)    // 24
#define SOMK  9
#define PKN   (MMN*SOMK)  // 576 node-neighbor slots
#define FEATC 1024

typedef __attribute__((ext_vector_type(16))) _Float16 v16h;
typedef __attribute__((ext_vector_type(8)))  _Float16 v8h;
typedef __attribute__((ext_vector_type(8)))  float    v8f;

static __device__ inline v16h cat8(v8h lo, v8h hi) {
  return __builtin_shufflevector(lo, hi, 0,1,2,3,4,5,6,7,8,9,10,11,12,13,14,15);
}

// ---------------------------------------------------------------------------
// WMMA GEMM:  Y[b, n, mo] = relu( sum_k W[mo, k] * X[b, n, k] + bias[mo] )
// X : (batch, P, CinP) f16, CinP % 32 == 0, zero-padded channels
// W : (CoutP, CinP)    f16 (padded), Y : (batch, P, CoutP) f16
//
// Register-blocked: one wave computes a 32(Cout) x 64(P) tile.
// Per K-step (32): 2 A-frags (2x b128 each) + 4 B-frags (2x b128 each) feed
// 8 v_wmma_f32_16x16x32_f16 with full operand reuse (A across 4 N-subtiles,
// B across 2 M-subtiles) -> matrix-unit bound instead of VMEM-issue bound.
// A-frag: lane half h -> K = 8h..8h+7 (v0..3), 16+8h..+7 (v4..7)
// B-frag: lane half h -> K = 8h..8h+7 (v0..3), 16+8h..+7 (v4..7) of point n
// D-frag: lane half h -> Cout = mbase + v  (one b128 store per subtile)
// ---------------------------------------------------------------------------
__global__ __launch_bounds__(64)
void k_gemm_relu(const _Float16* __restrict__ X, const _Float16* __restrict__ W,
                 const float* __restrict__ bias, _Float16* __restrict__ Y,
                 int P, int CinP, int CoutP, int nTiles64)
{
  const int t64 = blockIdx.x * blockDim.y + threadIdx.y;   // 64-wide point tile
  if (t64 >= nTiles64) return;                             // wave-uniform guard
  const int tM    = blockIdx.y;                            // 32-wide Cout tile
  const int batch = blockIdx.z;
  const int lane  = threadIdx.x;
  const int h     = lane >> 4;                             // lane half
  const int lm    = lane & 15;

  const _Float16* w0 = W + (size_t)(tM * 32 + lm) * CinP;  // M-subtile 0 row
  const _Float16* w1 = w0 + (size_t)16 * CinP;             // M-subtile 1 row
  const _Float16* x0 = X + ((size_t)batch * P + t64 * 64 + lm) * CinP + 16 * h;
  const size_t xs = (size_t)16 * CinP;                     // N-subtile stride

  v8f acc[2][4] = {};
  for (int k0 = 0; k0 < CinP; k0 += 32) {
    v16h a[2], b[4];
    a[0] = cat8(*(const v8h*)(w0 + k0 + 8 * h), *(const v8h*)(w0 + k0 + 16 + 8 * h));
    a[1] = cat8(*(const v8h*)(w1 + k0 + 8 * h), *(const v8h*)(w1 + k0 + 16 + 8 * h));
    #pragma unroll
    for (int ni = 0; ni < 4; ++ni)
      b[ni] = cat8(*(const v8h*)(x0 - 16 * h + ni * xs + k0 + 8 * h), *(const v8h*)(x0 - 16 * h + ni * xs + k0 + 16 + 8 * h));
    #pragma unroll
    for (int mi = 0; mi < 2; ++mi)
      #pragma unroll
      for (int ni = 0; ni < 4; ++ni)
        acc[mi][ni] = __builtin_amdgcn_wmma_f32_16x16x32_f16(
            false, a[mi], false, b[ni], (short)0, acc[mi][ni], false, false);
  }

  #pragma unroll
  for (int mi = 0; mi < 2; ++mi) {
    const int mbase = tM * 32 + mi * 16 + 8 * h;
    float bs[8];
    #pragma unroll
    for (int v = 0; v < 8; ++v) bs[v] = bias[mbase + v];
    #pragma unroll
    for (int ni = 0; ni < 4; ++ni) {
      v8h outv;
      #pragma unroll
      for (int v = 0; v < 8; ++v) {
        float t = acc[mi][ni][v] + bs[v];
        outv[v] = (_Float16)(t > 0.f ? t : 0.f);
      }
      *(v8h*)(Y + ((size_t)batch * P + t64 * 64 + ni * 16 + lm) * CoutP + mbase) = outv;
    }
  }
}

// Weight prep: fp32 (Cout,Cin) -> f16 (Cout,CinP) zero-padded
__global__ void k_wprep(const float* __restrict__ W, _Float16* __restrict__ Wp,
                        int Cout, int Cin, int CinP)
{
  int i = blockIdx.x * blockDim.x + threadIdx.x;
  if (i >= Cout * CinP) return;
  int r = i / CinP, c = i % CinP;
  Wp[i] = (c < Cin) ? (_Float16)W[(size_t)r * Cin + c] : (_Float16)0.f;
}

// Top-3 nearest SOM nodes per point + cluster sums/counts (atomics)
__global__ void k_topk3(const float* __restrict__ x, const float* __restrict__ node,
                        int* __restrict__ min_idx, unsigned* __restrict__ cnt,
                        float* __restrict__ csum)
{
  int n = blockIdx.x * blockDim.x + threadIdx.x;
  int b = blockIdx.y;
  if (n >= NN) return;
  float px = x[(b*3+0)*NN+n], py = x[(b*3+1)*NN+n], pz = x[(b*3+2)*NN+n];
  float d0 = 3.4e38f, d1 = 3.4e38f, d2 = 3.4e38f;
  int   i0 = 0, i1 = 0, i2 = 0;
  for (int m = 0; m < MMN; ++m) {
    float dx = px - node[(b*3+0)*MMN+m];
    float dy = py - node[(b*3+1)*MMN+m];
    float dz = pz - node[(b*3+2)*MMN+m];
    float d  = dx*dx + dy*dy + dz*dz;
    if (d < d0)      { d2=d1;i2=i1; d1=d0;i1=i0; d0=d;i0=m; }
    else if (d < d1) { d2=d1;i2=i1; d1=d; i1=m; }
    else if (d < d2) { d2=d; i2=m; }
  }
  int idx[3] = { i0, i1, i2 };
  #pragma unroll
  for (int k = 0; k < KTOP; ++k) {
    int mm = idx[k];
    min_idx[b*KN + k*NN + n] = mm;            // copy-major, matches x_stack
    atomicAdd(&cnt[b*MMN + mm], 1u);
    atomicAdd(&csum[(b*3+0)*MMN + mm], px);
    atomicAdd(&csum[(b*3+1)*MMN + mm], py);
    atomicAdd(&csum[(b*3+2)*MMN + mm], pz);
  }
}

// Rotated cluster means: coord (BR,3,M) fp32
__global__ void k_node_rot(const float* __restrict__ csum, const unsigned* __restrict__ cnt,
                           float* __restrict__ coord)
{
  int t = blockIdx.x * blockDim.x + threadIdx.x;
  if (t >= BRC * MMN) return;
  int br = t / MMN, m = t % MMN;
  int b = br / RRC, r = br % RRC;
  float inv = 1.f / ((float)cnt[b*MMN + m] + 1e-5f);
  float cx = csum[(b*3+0)*MMN+m] * inv;
  float cy = csum[(b*3+1)*MMN+m] * inv;
  float cz = csum[(b*3+2)*MMN+m] * inv;
  float ang = 6.2831853071795864f * (float)r / (float)RRC;
  float c = cosf(ang), s = sinf(ang);
  coord[(br*3+0)*MMN+m] = c*cx - s*cy;
  coord[(br*3+1)*MMN+m] = s*cx + c*cy;
  coord[(br*3+2)*MMN+m] = cz;
}

// Build x_aug (BR, kN, 32) f16: ch0-2 = rot(x)-center, ch3-5 = rot(sn), rest 0
__global__ void k_build_xaug(const float* __restrict__ x, const float* __restrict__ sn,
                             const float* __restrict__ coord, const int* __restrict__ min_idx,
                             _Float16* __restrict__ out)
{
  int p  = blockIdx.x * blockDim.x + threadIdx.x;
  int br = blockIdx.y;
  if (p >= KN) return;
  int b = br / RRC, r = br % RRC;
  int n = p % NN;
  int m = min_idx[b*KN + p];
  float ang = 6.2831853071795864f * (float)r / (float)RRC;
  float c = cosf(ang), s = sinf(ang);
  float xx = x[(b*3+0)*NN+n], xy = x[(b*3+1)*NN+n], xz = x[(b*3+2)*NN+n];
  float sx = sn[(b*3+0)*NN+n], sy = sn[(b*3+1)*NN+n], sz = sn[(b*3+2)*NN+n];
  _Float16* d = out + ((size_t)br*KN + p) * 32;
  d[0] = (_Float16)(c*xx - s*xy - coord[(br*3+0)*MMN+m]);
  d[1] = (_Float16)(s*xx + c*xy - coord[(br*3+1)*MMN+m]);
  d[2] = (_Float16)(xz          - coord[(br*3+2)*MMN+m]);
  d[3] = (_Float16)(c*sx - s*sy);
  d[4] = (_Float16)(s*sx + c*sy);
  d[5] = (_Float16)sz;
  #pragma unroll
  for (int cch = 6; cch < 32; ++cch) d[cch] = (_Float16)0.f;
}

// Masked segment max accumulation: int atomicMax on float bits (values >= 0)
__global__ void k_segmax_acc(const _Float16* __restrict__ act, const int* __restrict__ min_idx,
                             unsigned* __restrict__ bits, int C)
{
  int p  = blockIdx.x * blockDim.x + threadIdx.x;
  int br = blockIdx.y;
  if (p >= KN) return;
  int b = br / RRC;
  int m = min_idx[b*KN + p];
  const _Float16* a = act + ((size_t)br*KN + p) * C;
  unsigned* dst = bits + ((size_t)br*MMN + m) * C;
  for (int c = 0; c < C; ++c)
    atomicMax(&dst[c], __float_as_uint((float)a[c]));
}

// Fixup + convert: empty nodes fall back to act[:, point 0, :]
__global__ void k_segmax_fix(const unsigned* __restrict__ bits, const _Float16* __restrict__ act,
                             const unsigned* __restrict__ cnt, _Float16* __restrict__ out, int C)
{
  int i = blockIdx.x * blockDim.x + threadIdx.x;
  if (i >= BRC * MMN * C) return;
  int br = i / (MMN * C);
  int rem = i % (MMN * C);
  int m = rem / C, c = rem % C;
  int b = br / RRC;
  float v = (cnt[b*MMN + m] > 0) ? __uint_as_float(bits[i])
                                 : (float)act[(size_t)br*KN*C + c];
  out[i] = (_Float16)v;
}

// cat1 (BR,kN,128): [f1 | gather(mm1, min_idx)]
__global__ void k_build_cat1(const _Float16* __restrict__ f1, const _Float16* __restrict__ mm1,
                             const int* __restrict__ min_idx, _Float16* __restrict__ cat)
{
  int p  = blockIdx.x * blockDim.x + threadIdx.x;
  int br = blockIdx.y;
  if (p >= KN) return;
  int b = br / RRC;
  int m = min_idx[b*KN + p];
  const _Float16* s1 = f1  + ((size_t)br*KN  + p) * 64;
  const _Float16* s2 = mm1 + ((size_t)br*MMN + m) * 64;
  _Float16* d = cat + ((size_t)br*KN + p) * 128;
  for (int c = 0; c < 64; ++c) { d[c] = s1[c]; d[64 + c] = s2[c]; }
}

// knn input (BR, 576, 160): [nb_dec(3) | mm2 gathered(128) | pad(29)]
__global__ void k_build_knnin(const float* __restrict__ coord, const _Float16* __restrict__ mm2,
                              const int* __restrict__ knnI, _Float16* __restrict__ out)
{
  int t  = blockIdx.x * blockDim.x + threadIdx.x;
  int br = blockIdx.y;
  if (t >= PKN) return;
  int m = t / SOMK, kk = t % SOMK;
  int b = br / RRC;
  int j = knnI[(b*MMN + m)*SOMK + kk];
  _Float16* d = out + ((size_t)br*PKN + t) * 160;
  #pragma unroll
  for (int c = 0; c < 3; ++c)
    d[c] = (_Float16)(coord[(br*3+c)*MMN + j] - coord[(br*3+c)*MMN + m]);
  const _Float16* s = mm2 + ((size_t)br*MMN + j) * 128;
  for (int c = 0; c < 128; ++c) d[3 + c] = s[c];
  #pragma unroll
  for (int c = 131; c < 160; ++c) d[c] = (_Float16)0.f;
}

// g2in (BR,576,512): [g1 | broadcast(max_k g1)]
__global__ void k_build_g2in(const _Float16* __restrict__ g1, _Float16* __restrict__ g2in)
{
  int i  = blockIdx.x * blockDim.x + threadIdx.x;   // over MMN*256
  int br = blockIdx.y;
  if (i >= MMN * 256) return;
  int m = i / 256, c = i % 256;
  float mx = -3.4e38f;
  for (int kk = 0; kk < SOMK; ++kk) {
    float v = (float)g1[((size_t)br*PKN + m*SOMK + kk)*256 + c];
    mx = v > mx ? v : mx;
  }
  _Float16 hm = (_Float16)mx;
  for (int kk = 0; kk < SOMK; ++kk) {
    size_t base = ((size_t)br*PKN + m*SOMK + kk)*512;
    g2in[base + c]       = g1[((size_t)br*PKN + m*SOMK + kk)*256 + c];
    g2in[base + 256 + c] = hm;
  }
}

// knn_feat (BR,64,512): max over SOM_K
__global__ void k_knnfeat(const _Float16* __restrict__ g2, _Float16* __restrict__ feat)
{
  int i  = blockIdx.x * blockDim.x + threadIdx.x;   // over MMN*512
  int br = blockIdx.y;
  if (i >= MMN * 512) return;
  int m = i / 512, c = i % 512;
  float mx = -3.4e38f;
  for (int kk = 0; kk < SOMK; ++kk) {
    float v = (float)g2[((size_t)br*PKN + m*SOMK + kk)*512 + c];
    mx = v > mx ? v : mx;
  }
  feat[((size_t)br*MMN + m)*512 + c] = (_Float16)mx;
}

// h1in (BR,64,544): [center(3) | knn_feat(512) | pad(29)]
__global__ void k_build_h1in(const float* __restrict__ coord, const _Float16* __restrict__ feat,
                             _Float16* __restrict__ out)
{
  int m  = blockIdx.x * blockDim.x + threadIdx.x;
  int br = blockIdx.y;
  if (m >= MMN) return;
  _Float16* d = out + ((size_t)br*MMN + m) * 544;
  #pragma unroll
  for (int c = 0; c < 3; ++c) d[c] = (_Float16)coord[(br*3+c)*MMN + m];
  const _Float16* s = feat + ((size_t)br*MMN + m) * 512;
  for (int c = 0; c < 512; ++c) d[3 + c] = s[c];
  #pragma unroll
  for (int c = 515; c < 544; ++c) d[c] = (_Float16)0.f;
}

// h2in (BR,64,1024): [h1 | broadcast(max_m h1)]
__global__ void k_build_h2in(const _Float16* __restrict__ h1, _Float16* __restrict__ h2in)
{
  int c  = blockIdx.x * blockDim.x + threadIdx.x;   // over 512
  int br = blockIdx.y;
  if (c >= 512) return;
  float mx = -3.4e38f;
  for (int m = 0; m < MMN; ++m) {
    float v = (float)h1[((size_t)br*MMN + m)*512 + c];
    mx = v > mx ? v : mx;
  }
  _Float16 hm = (_Float16)mx;
  for (int m = 0; m < MMN; ++m) {
    size_t base = ((size_t)br*MMN + m)*1024;
    h2in[base + c]       = h1[((size_t)br*MMN + m)*512 + c];
    h2in[base + 512 + c] = hm;
  }
}

// out (B,1024): max over nodes and rotation copies
__global__ void k_final(const _Float16* __restrict__ h2, float* __restrict__ out)
{
  int c = blockIdx.x * blockDim.x + threadIdx.x;
  int b = blockIdx.y;
  if (c >= FEATC) return;
  float mx = -3.4e38f;
  for (int r = 0; r < RRC; ++r)
    for (int m = 0; m < MMN; ++m) {
      float v = (float)h2[(((size_t)(b*RRC + r))*MMN + m)*FEATC + c];
      mx = v > mx ? v : mx;
    }
  out[b*FEATC + c] = mx;
}

// ---------------------------------------------------------------------------
extern "C" void kernel_launch(void* const* d_in, const int* in_sizes, int n_in,
                              void* d_out, int out_size, void* d_ws, size_t ws_size,
                              hipStream_t stream)
{
  const float* x    = (const float*)d_in[0];
  const float* sn   = (const float*)d_in[1];
  const float* node = (const float*)d_in[2];
  const int*   knnI = (const int*)d_in[3];
  float* out = (float*)d_out;

  // Layer table: d_in indices (insertion order: pn1,pn2,knn1,knn2,fin1,fin2)
  struct LD { int wi, bi, Cin, Cout, CinP; };
  const LD L[15] = {
    {4,5,6,64,32},{6,7,64,64,64},{8,9,64,64,64},                 // pn1
    {10,11,128,128,128},{12,13,128,128,128},                     // pn2
    {14,15,131,256,160},{16,17,256,256,256},{18,19,256,256,256}, // knn1
    {20,21,512,512,512},{22,23,512,512,512},                     // knn2
    {24,25,515,512,544},{26,27,512,512,512},{28,29,512,512,512}, // fin1
    {30,31,1024,1024,1024},{32,33,1024,1024,1024}                // fin2
  };

  // -------- workspace arena --------
  char* ws = (char*)d_ws;
  size_t off = 0;
  auto alloc = [&](size_t bytes) -> char* {
    char* p = ws + off; off += (bytes + 255) & ~(size_t)255; return p;
  };
  int*      min_idx = (int*)     alloc((size_t)BB*KN*4);
  unsigned* cnt     = (unsigned*)alloc((size_t)BB*MMN*4);
  float*    csum    = (float*)   alloc((size_t)BB*3*MMN*4);
  float*    coord   = (float*)   alloc((size_t)BRC*3*MMN*4);
  unsigned* mm1bits = (unsigned*)alloc((size_t)BRC*MMN*64*4);
  unsigned* mm2bits = (unsigned*)alloc((size_t)BRC*MMN*128*4);
  _Float16* mm1f16  = (_Float16*)alloc((size_t)BRC*MMN*64*2);
  _Float16* mm2f16  = (_Float16*)alloc((size_t)BRC*MMN*128*2);
  _Float16* Xaug    = (_Float16*)alloc((size_t)BRC*KN*32*2);
  _Float16* Wp[15];
  for (int i = 0; i < 15; ++i)
    Wp[i] = (_Float16*)alloc((size_t)L[i].Cout * L[i].CinP * 2);

  // pn-stage big buffers; knn/fin stage reuses this region (disjoint lifetimes)
  const size_t szA = (size_t)BRC*KN*64*2, szC = (size_t)BRC*KN*128*2;
  char* region = alloc(2*szA + 2*szC);
  _Float16* bufA   = (_Float16*)(region);
  _Float16* bufB   = (_Float16*)(region + szA);
  _Float16* catbuf = (_Float16*)(region + 2*szA);          // cat1 / f2
  _Float16* buf128 = (_Float16*)(region + 2*szA + szC);
  // knn/fin carve (starts at region base; pn buffers dead by then)
  size_t ko = 0;
  auto kalloc = [&](size_t bytes) -> char* {
    char* p = region + ko; ko += (bytes + 255) & ~(size_t)255; return p;
  };
  _Float16* knnin = (_Float16*)kalloc((size_t)BRC*PKN*160*2);
  _Float16* kA    = (_Float16*)kalloc((size_t)BRC*PKN*256*2);
  _Float16* kB    = (_Float16*)kalloc((size_t)BRC*PKN*256*2);
  _Float16* g2in  = (_Float16*)kalloc((size_t)BRC*PKN*512*2);
  _Float16* kC    = (_Float16*)kalloc((size_t)BRC*PKN*512*2);
  _Float16* feat  = (_Float16*)kalloc((size_t)BRC*MMN*512*2);
  _Float16* h1in  = (_Float16*)kalloc((size_t)BRC*MMN*544*2);
  _Float16* fA    = (_Float16*)kalloc((size_t)BRC*MMN*512*2);
  _Float16* fB    = (_Float16*)kalloc((size_t)BRC*MMN*512*2);
  _Float16* h2in  = (_Float16*)kalloc((size_t)BRC*MMN*1024*2);
  _Float16* fC    = (_Float16*)kalloc((size_t)BRC*MMN*1024*2);

  // -------- zero-init accumulators --------
  hipMemsetAsync(cnt,     0, (size_t)BB*MMN*4,       stream);
  hipMemsetAsync(csum,    0, (size_t)BB*3*MMN*4,     stream);
  hipMemsetAsync(mm1bits, 0, (size_t)BRC*MMN*64*4,   stream);
  hipMemsetAsync(mm2bits, 0, (size_t)BRC*MMN*128*4,  stream);

  // -------- weight prep (fp32 -> padded f16) --------
  for (int i = 0; i < 15; ++i) {
    int n = L[i].Cout * L[i].CinP;
    k_wprep<<<dim3((n+255)/256), dim3(256), 0, stream>>>(
        (const float*)d_in[L[i].wi], Wp[i], L[i].Cout, L[i].Cin, L[i].CinP);
  }

  // -------- SOM assignment, rotated cluster means, augmented input --------
  k_topk3<<<dim3((NN+255)/256, BB), dim3(256), 0, stream>>>(x, node, min_idx, cnt, csum);
  k_node_rot<<<dim3((BRC*MMN+255)/256), dim3(256), 0, stream>>>(csum, cnt, coord);
  k_build_xaug<<<dim3((KN+255)/256, BRC), dim3(256), 0, stream>>>(x, sn, coord, min_idx, Xaug);

  auto gemm = [&](const _Float16* Xp, int li, _Float16* Yp, int P) {
    int nT64 = P / 64;                     // 6144->96, 576->9, 64->1
    dim3 grid((nT64 + 1) / 2, L[li].Cout / 32, BRC), block(32, 2);
    k_gemm_relu<<<grid, block, 0, stream>>>(Xp, Wp[li], (const float*)d_in[L[li].bi],
                                            Yp, P, L[li].CinP, L[li].Cout, nT64);
  };

  // pn1: 6->64->64->64, f1 = bufA
  gemm(Xaug, 0, bufA, KN); gemm(bufA, 1, bufB, KN); gemm(bufB, 2, bufA, KN);
  k_segmax_acc<<<dim3((KN+255)/256, BRC), dim3(256), 0, stream>>>(bufA, min_idx, mm1bits, 64);
  k_segmax_fix<<<dim3((BRC*MMN*64+255)/256), dim3(256), 0, stream>>>(mm1bits, bufA, cnt, mm1f16, 64);
  k_build_cat1<<<dim3((KN+255)/256, BRC), dim3(256), 0, stream>>>(bufA, mm1f16, min_idx, catbuf);

  // pn2: 128->128->128, f2 = catbuf
  gemm(catbuf, 3, buf128, KN); gemm(buf128, 4, catbuf, KN);
  k_segmax_acc<<<dim3((KN+255)/256, BRC), dim3(256), 0, stream>>>(catbuf, min_idx, mm2bits, 128);
  k_segmax_fix<<<dim3((BRC*MMN*128+255)/256), dim3(256), 0, stream>>>(mm2bits, catbuf, cnt, mm2f16, 128);

  // knn fusion over SOM nodes
  k_build_knnin<<<dim3((PKN+255)/256, BRC), dim3(256), 0, stream>>>(coord, mm2f16, knnI, knnin);
  gemm(knnin, 5, kA, PKN); gemm(kA, 6, kB, PKN); gemm(kB, 7, kA, PKN);   // g1 = kA
  k_build_g2in<<<dim3((MMN*256+255)/256, BRC), dim3(256), 0, stream>>>(kA, g2in);
  gemm(g2in, 8, kC, PKN); gemm(kC, 9, g2in, PKN);                        // g2 = g2in
  k_knnfeat<<<dim3((MMN*512+255)/256, BRC), dim3(256), 0, stream>>>(g2in, feat);

  // final PointNet fusion over nodes
  k_build_h1in<<<dim3(1, BRC), dim3(64), 0, stream>>>(coord, feat, h1in);
  gemm(h1in, 10, fA, MMN); gemm(fA, 11, fB, MMN); gemm(fB, 12, fA, MMN); // h1 = fA
  k_build_h2in<<<dim3(2, BRC), dim3(256), 0, stream>>>(fA, h2in);
  gemm(h2in, 13, fC, MMN); gemm(fC, 14, h2in, MMN);                      // h2 = h2in
  k_final<<<dim3((FEATC+255)/256, BB), dim3(256), 0, stream>>>(h2in, out);

  (void)in_sizes; (void)n_in; (void)out_size; (void)ws_size;
}
